// GPT_84335977824372
// MI455X (gfx1250) — compile-verified, matrix operand order checked by reading, not run on hardware
//
#include <hip/hip_runtime.h>
#include <hip/hip_bf16.h>

#define DIMS   256
#define VOCAB  4096
#define LAYERS 6
#define HEADS  2
#define BB     2
#define SS     1024
#define DH     128

typedef __attribute__((ext_vector_type(16))) _Float16 v16h;
typedef __attribute__((ext_vector_type(8)))  _Float16 v8h;
typedef __attribute__((ext_vector_type(8)))  float    v8f;
typedef __attribute__((ext_vector_type(4)))  unsigned int v4u;
typedef __attribute__((ext_vector_type(8)))  int v8i;
typedef __attribute__((ext_vector_type(4)))  int v4i;

// ---------------------------------------------------------------------------
// CDNA5 16-bit A/B fragment (16x32 / 32x16): lane L -> row = L%16, half = L/16;
// per lane the fragment is two contiguous 16-byte K-chunks at K = 8*half and
// K = 16 + 8*half.  LDS tiles are K-contiguous with row stride 40 halfs (80 B):
// 16B-aligned b128 reads, bank-conflict-free, and reproducible by the TDM's
// pad feature (pad 4 DWORDs every 16 DWORDs).
// ---------------------------------------------------------------------------
__device__ __forceinline__ v16h lds_frag(const _Float16* p0, int stride, int row,
                                         int koff, int lane) {
  const int hf = lane >> 4;
  const _Float16* p = p0 + row * stride + koff + 8 * hf;
  v8h lo = *(const v8h*)(p);
  v8h hi = *(const v8h*)(p + 16);
  v16h r;
#pragma unroll
  for (int i = 0; i < 8; ++i) { r[i] = lo[i]; r[i + 8] = hi[i]; }
  return r;
}

// Fragment straight from global f16 (row-major, K-contiguous, 16B aligned).
__device__ __forceinline__ v16h glb_frag(const _Float16* rowp, int koff, int lane) {
  const int hf = lane >> 4;
  const _Float16* p = rowp + koff + 8 * hf;
  v8h lo = *(const v8h*)(p);
  v8h hi = *(const v8h*)(p + 16);
  v16h r;
#pragma unroll
  for (int i = 0; i < 8; ++i) { r[i] = lo[i]; r[i + 8] = hi[i]; }
  return r;
}

__device__ __forceinline__ v8f wmma_f16(v16h a, v16h b, v8f c) {
  return __builtin_amdgcn_wmma_f32_16x16x32_f16(false, a, false, b, (short)0, c,
                                                false, false);
}

// ---------------------------------------------------------------------------
// TDM descriptor builders (D# group0 / group1) for a 2-D f16 tile:
//   tile = 64 rows x 32 halfs from a [rows, K] row-major f16 tensor,
//   written to LDS with 16 B of padding after every 64 B (-> 40-half rows).
// All inputs are wave-uniform (kernargs / blockIdx / loop counter).
// ---------------------------------------------------------------------------
__device__ __forceinline__ v4u tdm_g0(unsigned lds_byte_addr, const _Float16* gaddr) {
  unsigned long long ga = (unsigned long long)gaddr;
  v4u g;
  g[0] = 1u;                                   // count=1, user descriptor
  g[1] = lds_byte_addr;                        // lds_addr
  g[2] = (unsigned)ga;                         // global_addr[31:0]
  g[3] = (unsigned)(ga >> 32) | (2u << 30);    // global_addr[56:32] | type=2
  return g;
}

__device__ __forceinline__ v8i tdm_g1(int K, int rows) {
  v8i g;
  // data_size=1 (2B) | pad_enable | pad_interval=3 (16 DW) | pad_amount=3 (4 DW)
  g[0] = (1 << 16) | (1 << 20) | (3 << 22) | (3 << 25);
  g[1] = (K & 0xFFFF) << 16;                   // tensor_dim0[15:0]
  g[2] = (int)(((unsigned)K >> 16) | (((unsigned)rows & 0xFFFF) << 16));
  g[3] = (int)(((unsigned)rows >> 16) | (32u << 16));  // tensor_dim1 hi | tile_dim0=32
  g[4] = 64;                                   // tile_dim1=64, tile_dim2=0
  g[5] = K;                                    // tensor_dim0_stride lo
  g[6] = 0;
  g[7] = 0;
  return g;
}

// ---------------------------------------------------------------------------
// Weight convert + transpose: W[K,N] f32 -> Wt[N,K] f16 (runs once, ~8 MB).
// ---------------------------------------------------------------------------
__global__ void wconv_kernel(const float* __restrict__ W, _Float16* __restrict__ Wt,
                             int K, int N) {
  const int idx = blockIdx.x * 256 + threadIdx.x;
  const int k = idx / N, n = idx % N;
  Wt[(size_t)n * K + k] = (_Float16)W[idx];
}

// ---------------------------------------------------------------------------
// Encoder: x[b,s,o] = dot(W_enc[o, X[b,s], :], pos[s,:]) + b_enc[o]
// Pure HBM-bandwidth kernel (512 MB of W_enc gather dominates the whole net).
// Writes the f32 residual stream and its f16 shadow (WMMA operand).
// ---------------------------------------------------------------------------
__global__ void encode_kernel(const int* __restrict__ X,
                              const float* __restrict__ Wenc,
                              const float* __restrict__ benc,
                              float* __restrict__ xout,
                              _Float16* __restrict__ xh) {
  __shared__ float pos[DIMS];
  const int idx = blockIdx.x;            // token index in [0, B*S)
  const int s = idx & (SS - 1);
  const int t = threadIdx.x;             // output dim
  const float p = (float)BB - (float)s;  // faithful to reference's odd indexing
  {
    int j = (t < 128) ? t : (t - 128);
    float invf = __expf(-((float)j / 128.0f) * 9.210340372f);  // 10000^(-j/128)
    float ang = p * invf;
    pos[t] = (t < 128) ? __sinf(ang) : __cosf(ang);
  }
  __syncthreads();
  const int tok = X[idx];
  const float* w = Wenc + ((size_t)t * VOCAB + tok) * DIMS;
  float acc = 0.0f;
#pragma unroll 4
  for (int j = 0; j < DIMS; j += 4) {
    float4 wv = *(const float4*)(w + j);
    acc += wv.x * pos[j] + wv.y * pos[j + 1] + wv.z * pos[j + 2] + wv.w * pos[j + 3];
  }
  const float y = acc + benc[t];
  xout[(size_t)idx * DIMS + t] = y;
  xh[(size_t)idx * DIMS + t] = (_Float16)y;
}

// ---------------------------------------------------------------------------
// TDM + WMMA GEMM: C[M,N] = epi(A[M,K] @ Bt[N,K]^T), A/Bt f16, K-contiguous.
// Block: 256 threads (8 waves), 64x64 tile, K-step 32.
// Wave 0 issues two tensor_load_to_lds (A tile, B tile) -> LDS (TDM applies
// the 40-half row padding), waits TENSORcnt, barrier, everyone runs WMMA.
// EPI: 0 = f16 store, 1 = SiLU f16 store, 2 = residual f32 + f16 shadow,
//      3 = plain f32 store (decoder).
// ---------------------------------------------------------------------------
template <int EPI>
__global__ void gemm_kernel(const _Float16* __restrict__ A,
                            const _Float16* __restrict__ Bt16,
                            _Float16* __restrict__ Ch, float* __restrict__ Cf,
                            const float* Res, int M, int N, int K) {
  __shared__ _Float16 At[64 * 40];
  __shared__ _Float16 Bt[64 * 40];
  const int tid = threadIdx.x;
  const int lane = tid & 31, wid = tid >> 5;
  const int m0 = blockIdx.y * 64, n0 = blockIdx.x * 64;
  const int r = wid & 3, c0 = (wid >> 2) * 2;
  v8f acc0 = {};
  v8f acc1 = {};
  const unsigned at_lds = (unsigned)(size_t)&At[0];
  const unsigned bt_lds = (unsigned)(size_t)&Bt[0];
  const v8i g1a = tdm_g1(K, M);
  const v8i g1b = tdm_g1(K, N);
  const v4i z4 = {};
  const v8i z8 = {};
  for (int k0 = 0; k0 < K; k0 += 32) {
    if (wid == 0) {
      v4u g0a = tdm_g0(at_lds, A + (size_t)m0 * K + k0);
      v4u g0b = tdm_g0(bt_lds, Bt16 + (size_t)n0 * K + k0);
      __builtin_amdgcn_tensor_load_to_lds(g0a, g1a, z4, z4, z8, 0);
      __builtin_amdgcn_tensor_load_to_lds(g0b, g1b, z4, z4, z8, 0);
      __builtin_amdgcn_s_wait_tensorcnt(0);
    }
    __syncthreads();
    v16h af  = lds_frag(At, 40, r * 16 + (lane & 15), 0, lane);
    v16h bf0 = lds_frag(Bt, 40, c0 * 16 + (lane & 15), 0, lane);
    v16h bf1 = lds_frag(Bt, 40, (c0 + 1) * 16 + (lane & 15), 0, lane);
    acc0 = wmma_f16(af, bf0, acc0);
    acc1 = wmma_f16(af, bf1, acc1);
    __syncthreads();
  }
  // D layout: VGPR i -> M = i + 8*(lane/16), N = lane%16
  const int mbase = m0 + r * 16 + 8 * (lane >> 4);
  const int nA = n0 + c0 * 16 + (lane & 15);
#pragma unroll
  for (int i = 0; i < 8; ++i) {
    const int mm = mbase + i;
    float y0 = acc0[i], y1 = acc1[i];
    if (EPI == 1) {
      y0 = y0 / (1.0f + __expf(-y0));
      y1 = y1 / (1.0f + __expf(-y1));
    }
    if (EPI == 2) {
      y0 += Res[(size_t)mm * N + nA];
      y1 += Res[(size_t)mm * N + nA + 16];
      Cf[(size_t)mm * N + nA] = y0;
      Cf[(size_t)mm * N + nA + 16] = y1;
      Ch[(size_t)mm * N + nA] = (_Float16)y0;
      Ch[(size_t)mm * N + nA + 16] = (_Float16)y1;
    } else if (EPI == 3) {
      Cf[(size_t)mm * N + nA] = y0;
      Cf[(size_t)mm * N + nA + 16] = y1;
    } else {
      Ch[(size_t)mm * N + nA] = (_Float16)y0;
      Ch[(size_t)mm * N + nA + 16] = (_Float16)y1;
    }
  }
}

// ---------------------------------------------------------------------------
// Causal attention, one block per (b, head, 16 q-rows), 8 waves.
// q/k/v arrive as f16 (QKV GEMM output) -> fragments are direct b128 loads.
// 1/sqrt(dh) applied to f32 score accumulators.  Softmax in LDS f32,
// probabilities stored f16, P@V via WMMA.  Dynamic LDS = 98816 B (< 320 KB).
// ---------------------------------------------------------------------------
#define SC_STRIDE 1024
#define PM_STRIDE 1040
#define ATTN_SMEM (16 * SC_STRIDE * 4 + 16 * PM_STRIDE * 2)

__global__ void attn_kernel(const _Float16* __restrict__ Qh,
                            const _Float16* __restrict__ Kh,
                            const _Float16* __restrict__ Vh,
                            _Float16* __restrict__ Oh) {
  extern __shared__ char smem[];
  float* sc = (float*)smem;
  _Float16* pm = (_Float16*)(smem + 16 * SC_STRIDE * 4);
  const int tid = threadIdx.x, lane = tid & 31, wid = tid >> 5;
  const int hf = lane >> 4, l16 = lane & 15;
  const int q0 = blockIdx.x * 16, h = blockIdx.y, b = blockIdx.z;
  const float scale = 0.08838834764831845f;  // 1/sqrt(128)

  v16h qf[4];
  {
    const _Float16* qp = Qh + ((size_t)(b * SS + q0 + l16)) * DIMS + h * DH;
#pragma unroll
    for (int f = 0; f < 4; ++f) qf[f] = glb_frag(qp, f * 32, lane);
  }

  const int ntiles = (q0 >> 4) + 1;       // causal: only k-tiles <= q-tile
  const int nt2 = (ntiles + 1) & ~1;      // pad to K-step of 32 for the PV GEMM
  const int L = nt2 * 16;

  for (int t = wid; t < nt2; t += 8) {
    const int k0 = t * 16;
    v8f acc = {};
    const _Float16* kp = Kh + ((size_t)(b * SS + k0 + l16)) * DIMS + h * DH;
#pragma unroll
    for (int f = 0; f < 4; ++f) {
      v16h kf = glb_frag(kp, f * 32, lane);
      acc = wmma_f16(qf[f], kf, acc);
    }
#pragma unroll
    for (int i = 0; i < 8; ++i) {
      const int mrow = i + 8 * hf;
      const int kg = k0 + l16;
      sc[mrow * SC_STRIDE + kg] = (kg <= q0 + mrow) ? acc[i] * scale : -1e30f;
    }
  }
  __syncthreads();

  for (int row = wid * 2; row < wid * 2 + 2; ++row) {
    float mx = -3.4e38f;
    for (int j = lane; j < L; j += 32) mx = fmaxf(mx, sc[row * SC_STRIDE + j]);
#pragma unroll
    for (int o = 16; o > 0; o >>= 1) mx = fmaxf(mx, __shfl_xor(mx, o, 32));
    float sum = 0.0f;
    for (int j = lane; j < L; j += 32) {
      float e = __expf(sc[row * SC_STRIDE + j] - mx);
      sum += e;
      sc[row * SC_STRIDE + j] = e;
    }
#pragma unroll
    for (int o = 16; o > 0; o >>= 1) sum += __shfl_xor(sum, o, 32);
    const float inv = 1.0f / sum;
    for (int j = lane; j < L; j += 32)
      pm[row * PM_STRIDE + j] = (_Float16)(sc[row * SC_STRIDE + j] * inv);
  }
  __syncthreads();

  const int colg = h * DH + wid * 16 + l16;
  v8f acc = {};
  for (int ks = 0; ks < L; ks += 32) {
    v16h pf = lds_frag(pm, PM_STRIDE, l16, ks, lane);
    const _Float16* vp = Vh + ((size_t)(b * SS + ks + 8 * hf)) * DIMS + colg;
    v16h vf;
#pragma unroll
    for (int i = 0; i < 8; ++i) {
      vf[i] = vp[(size_t)i * DIMS];
      vf[i + 8] = vp[(size_t)(16 + i) * DIMS];
    }
    acc = wmma_f16(pf, vf, acc);
  }
#pragma unroll
  for (int i = 0; i < 8; ++i) {
    const int mrow = i + 8 * hf;
    Oh[((size_t)(b * SS + q0 + mrow)) * DIMS + colg] = (_Float16)acc[i];
  }
}

// ---------------------------------------------------------------------------
extern "C" void kernel_launch(void* const* d_in, const int* in_sizes, int n_in,
                              void* d_out, int out_size, void* d_ws, size_t ws_size,
                              hipStream_t stream) {
  (void)in_sizes; (void)n_in; (void)out_size; (void)ws_size;
  const int*   X    = (const int*)d_in[0];
  const float* Wenc = (const float*)d_in[1];
  const float* benc = (const float*)d_in[2];
  const float* Wq   = (const float*)d_in[3];
  const float* Wk   = (const float*)d_in[4];
  const float* Wv   = (const float*)d_in[5];
  const float* Wo   = (const float*)d_in[6];
  const float* W1   = (const float*)d_in[7];
  const float* W2   = (const float*)d_in[8];
  const float* Wdec = (const float*)d_in[9];
  float* out = (float*)d_out;

  const int T = BB * SS;  // 2048 tokens
  const size_t TD = (size_t)T * DIMS;

  float* x = (float*)d_ws;                     // [T,256] f32 residual stream
  _Float16* hb = (_Float16*)(x + TD);          // f16 region
  _Float16* xh   = hb;                         // [T,256] residual shadow
  _Float16* qh   = xh + TD;
  _Float16* kh   = qh + TD;
  _Float16* vh   = kh + TD;
  _Float16* oh   = vh + TD;                    // attention out (pre-Wo)
  _Float16* ah   = oh + TD;                    // o @ Wo (MLP input)
  _Float16* h1h  = ah + TD;                    // [T,512] MLP hidden
  _Float16* wqt  = h1h + (size_t)T * 2 * DIMS; // transposed f16 weights
  _Float16* wkt  = wqt + (size_t)LAYERS * DIMS * DIMS;
  _Float16* wvt  = wkt + (size_t)LAYERS * DIMS * DIMS;
  _Float16* wot  = wvt + (size_t)LAYERS * DIMS * DIMS;
  _Float16* w1t  = wot + (size_t)LAYERS * DIMS * DIMS;
  _Float16* w2t  = w1t + (size_t)LAYERS * DIMS * 2 * DIMS;
  _Float16* wdt  = w2t + (size_t)LAYERS * 2 * DIMS * DIMS;

  // one-time weight convert+transpose (f32 [K,N] -> f16 [N,K])
  for (int l = 0; l < LAYERS; ++l) {
    const size_t o1 = (size_t)l * DIMS * DIMS, o2 = (size_t)l * DIMS * 2 * DIMS;
    wconv_kernel<<<dim3(DIMS * DIMS / 256), 256, 0, stream>>>(Wq + o1, wqt + o1, DIMS, DIMS);
    wconv_kernel<<<dim3(DIMS * DIMS / 256), 256, 0, stream>>>(Wk + o1, wkt + o1, DIMS, DIMS);
    wconv_kernel<<<dim3(DIMS * DIMS / 256), 256, 0, stream>>>(Wv + o1, wvt + o1, DIMS, DIMS);
    wconv_kernel<<<dim3(DIMS * DIMS / 256), 256, 0, stream>>>(Wo + o1, wot + o1, DIMS, DIMS);
    wconv_kernel<<<dim3(DIMS * 2 * DIMS / 256), 256, 0, stream>>>(W1 + o2, w1t + o2, DIMS, 2 * DIMS);
    wconv_kernel<<<dim3(DIMS * 2 * DIMS / 256), 256, 0, stream>>>(W2 + o2, w2t + o2, 2 * DIMS, DIMS);
  }
  wconv_kernel<<<dim3(DIMS * VOCAB / 256), 256, 0, stream>>>(Wdec, wdt, DIMS, VOCAB);

  encode_kernel<<<dim3(T), 256, 0, stream>>>(X, Wenc, benc, x, xh);

  const dim3 g256(DIMS / 64, T / 64);
  const dim3 g512(2 * DIMS / 64, T / 64);
  for (int l = 0; l < LAYERS; ++l) {
    const size_t o1 = (size_t)l * DIMS * DIMS, o2 = (size_t)l * DIMS * 2 * DIMS;
    gemm_kernel<0><<<g256, 256, 0, stream>>>(xh, wqt + o1, qh, nullptr, nullptr, T, DIMS, DIMS);
    gemm_kernel<0><<<g256, 256, 0, stream>>>(xh, wkt + o1, kh, nullptr, nullptr, T, DIMS, DIMS);
    gemm_kernel<0><<<g256, 256, 0, stream>>>(xh, wvt + o1, vh, nullptr, nullptr, T, DIMS, DIMS);
    attn_kernel<<<dim3(SS / 16, HEADS, BB), 256, ATTN_SMEM, stream>>>(qh, kh, vh, oh);
    gemm_kernel<0><<<g256, 256, 0, stream>>>(oh, wot + o1, ah, nullptr, nullptr, T, DIMS, DIMS);
    gemm_kernel<1><<<g512, 256, 0, stream>>>(ah, w1t + o2, h1h, nullptr, nullptr, T, 2 * DIMS, DIMS);
    gemm_kernel<2><<<g256, 256, 0, stream>>>(h1h, w2t + o2, xh, x, x, T, DIMS, 2 * DIMS);
  }
  gemm_kernel<3><<<dim3(VOCAB / 64, T / 64), 256, 0, stream>>>(xh, wdt, nullptr, out,
                                                               nullptr, T, VOCAB, DIMS);
}
